// CrossModalAttention_54485955117599
// MI455X (gfx1250) — matrix operand order repaired and checked by reading, not compiled, not run on hardware
//
#include <hip/hip_runtime.h>
#include <hip/hip_bf16.h>

// CrossModalAttention for MI455X (gfx1250, wave32, WMMA f16 16x16x32, f32 accum)
// B=8, NQ=NK=2048, D=512
#define BATCH 8
#define NQ    2048
#define NKEY  2048
#define DDIM  512

typedef __attribute__((ext_vector_type(16))) _Float16 v16h;
typedef __attribute__((ext_vector_type(8)))  _Float16 h8;     // 16 bytes
typedef __attribute__((ext_vector_type(8)))  float    v8f;
typedef __attribute__((ext_vector_type(4)))  float    f32x4;

union F16Frag { v16h v; h8 h[2]; _Float16 e[16]; };

// Reuse hints must be integer constant expressions -> template params.
template <bool RA = false, bool RB = false>
static __device__ __forceinline__ v8f wmma_f16(const F16Frag& a, const F16Frag& b, v8f c) {
    return __builtin_amdgcn_wmma_f32_16x16x32_f16(false, a.v, false, b.v,
                                                  (short)0, c, RA, RB);
}

// ---------------------------------------------------------------------------
// Projection: Y[m,e] = sum_d X[m,d]*W[e,d] + bias[e], converted to f16.
// VT=0: Y stored [m][e] row-major (m = b*2048+n).
// VT=1: Y stored per-batch transposed: Y[((b*D)+e)*NK + n]  (for V).
// Block = 256 thr (8 waves). Tile = 128 rows x 64 cols, K chunk = 32.
// Double-buffered LDS (ping-pong) with register staging to overlap the f32
// global traffic (dominant HBM phase) with the WMMAs.
// ---------------------------------------------------------------------------
template <int VT>
__global__ __launch_bounds__(256)
void proj_f16_kernel(const float* __restrict__ X, const float* __restrict__ W,
                     const float* __restrict__ bias, _Float16* __restrict__ Y)
{
    const int tid   = threadIdx.x;
    const int wave  = tid >> 5;
    const int lane  = tid & 31;
    const int lrow  = lane & 15;
    const int hisel = lane >> 4;          // 0: lanes 0-15, 1: lanes 16-31
    const int m0    = blockIdx.x * 128;
    const int n0    = blockIdx.y * 64;

    __shared__ __attribute__((aligned(16))) _Float16 As[2][128][40]; // 128x32 (+pad)
    __shared__ __attribute__((aligned(16))) _Float16 Bs[2][64][40];  // 64x32 (+pad)

    v8f acc[4];
    #pragma unroll
    for (int ct = 0; ct < 4; ++ct)
        for (int j = 0; j < 8; ++j) acc[ct][j] = 0.0f;

    // register staging for one K-chunk
    f32x4 xr[4], wr[2];

    // per-thread staging coordinates (fixed across chunks)
    int xrow[4], xcol[4], wrow[2], wcol[2];
    #pragma unroll
    for (int it = 0; it < 4; ++it) {
        int idx = (it * 256 + tid) * 4;
        xrow[it] = idx >> 5; xcol[it] = idx & 31;
    }
    #pragma unroll
    for (int it = 0; it < 2; ++it) {
        int idx = (it * 256 + tid) * 4;
        wrow[it] = idx >> 5; wcol[it] = idx & 31;
    }

    auto load_chunk = [&](int kc) {
        #pragma unroll
        for (int it = 0; it < 4; ++it)
            xr[it] = *(const f32x4*)(X + (size_t)(m0 + xrow[it]) * DDIM + kc + xcol[it]);
        #pragma unroll
        for (int it = 0; it < 2; ++it)
            wr[it] = *(const f32x4*)(W + (size_t)(n0 + wrow[it]) * DDIM + kc + wcol[it]);
    };
    auto store_chunk = [&](int buf) {
        #pragma unroll
        for (int it = 0; it < 4; ++it) {
            As[buf][xrow[it]][xcol[it] + 0] = (_Float16)xr[it].x;
            As[buf][xrow[it]][xcol[it] + 1] = (_Float16)xr[it].y;
            As[buf][xrow[it]][xcol[it] + 2] = (_Float16)xr[it].z;
            As[buf][xrow[it]][xcol[it] + 3] = (_Float16)xr[it].w;
        }
        #pragma unroll
        for (int it = 0; it < 2; ++it) {
            Bs[buf][wrow[it]][wcol[it] + 0] = (_Float16)wr[it].x;
            Bs[buf][wrow[it]][wcol[it] + 1] = (_Float16)wr[it].y;
            Bs[buf][wrow[it]][wcol[it] + 2] = (_Float16)wr[it].z;
            Bs[buf][wrow[it]][wcol[it] + 3] = (_Float16)wr[it].w;
        }
    };

    load_chunk(0);
    store_chunk(0);
    __syncthreads();

    #pragma unroll 4
    for (int i = 0; i < DDIM / 32; ++i) {
        const int cur = i & 1;
        // issue next chunk's global loads before touching LDS (overlap)
        if (i < DDIM / 32 - 1) load_chunk((i + 1) * 32);

        // fragments: A shared by 4 WMMAs; all B fragments in distinct regs
        F16Frag a;
        {
            const _Float16* ap = &As[cur][wave * 16 + lrow][hisel * 16];
            a.h[0] = *(const h8*)(ap);
            a.h[1] = *(const h8*)(ap + 8);
        }
        F16Frag bf[4];
        #pragma unroll
        for (int ct = 0; ct < 4; ++ct) {
            const _Float16* bp = &Bs[cur][ct * 16 + lrow][hisel * 16];
            bf[ct].h[0] = *(const h8*)(bp);
            bf[ct].h[1] = *(const h8*)(bp + 8);
        }
        // back-to-back WMMAs, A-operand reuse hint on all but the last
        acc[0] = wmma_f16<true,  false>(a, bf[0], acc[0]);
        acc[1] = wmma_f16<true,  false>(a, bf[1], acc[1]);
        acc[2] = wmma_f16<true,  false>(a, bf[2], acc[2]);
        acc[3] = wmma_f16<false, false>(a, bf[3], acc[3]);

        if (i < DDIM / 32 - 1) store_chunk(cur ^ 1);
        __syncthreads();
    }

    // Epilogue: bias add, f16 convert, store (compile-time layout)
    #pragma unroll
    for (int ct = 0; ct < 4; ++ct) {
        const int e = n0 + ct * 16 + lrow;
        const float bv = bias[e];
        #pragma unroll
        for (int j = 0; j < 8; ++j) {
            int row = m0 + wave * 16 + hisel * 8 + j;
            float val = acc[ct][j] + bv;
            if (VT == 0) {
                Y[(size_t)row * DDIM + e] = (_Float16)val;
            } else {
                int b = row >> 11;        // / 2048
                int n = row & 2047;
                Y[((size_t)b * DDIM + e) * NKEY + n] = (_Float16)val;
            }
        }
    }
}

// ---------------------------------------------------------------------------
// Flash attention: block = (32 queries, one batch), 256 thr = 8 waves.
// Per 64-key block: each wave computes one 16x16 S tile (K=512 -> 16 WMMAs),
// online softmax in LDS, then P(LDS f16) @ V(global, transposed) with each
// wave owning a 64-wide slice of the D=512 output dims. K/V are L2-resident
// (4 MB/batch vs 192 MB L2); prefetch the next key block each iteration.
// ---------------------------------------------------------------------------
__global__ __launch_bounds__(256)
void attn_kernel(const _Float16* __restrict__ qh, const _Float16* __restrict__ kh,
                 const _Float16* __restrict__ vt, float* __restrict__ out)
{
    const int tid   = threadIdx.x;
    const int wave  = tid >> 5;
    const int lane  = tid & 31;
    const int lrow  = lane & 15;
    const int hisel = lane >> 4;

    const int q0 = blockIdx.x * 32;
    const int b  = blockIdx.y;

    const _Float16* qb = qh + (size_t)b * NQ   * DDIM;
    const _Float16* kb = kh + (size_t)b * NKEY * DDIM;
    const _Float16* vb = vt + (size_t)b * DDIM * NKEY;
    float*          ob = out + (size_t)b * NQ  * DDIM;

    __shared__ __attribute__((aligned(16))) float    S_lds[32][64];
    __shared__ __attribute__((aligned(16))) _Float16 P_lds[32][72]; // pad stride
    __shared__ float m_s[32], l_s[32], f_s[32];

    if (tid < 32) { m_s[tid] = -3.0e38f; l_s[tid] = 0.0f; }

    const int srw = wave >> 2;        // S row tile (0..1)
    const int scw = wave & 3;         // S col tile (0..3)
    const int cb  = wave * 64;        // this wave's output-dim slice

    v8f oacc[2][4];
    #pragma unroll
    for (int r = 0; r < 2; ++r)
        for (int ct = 0; ct < 4; ++ct)
            for (int j = 0; j < 8; ++j) oacc[r][ct][j] = 0.0f;

    const _Float16* qrow = qb + (size_t)(q0 + srw * 16 + lrow) * DDIM;

    for (int kb0 = 0; kb0 < NKEY; kb0 += 64) {
        // prefetch next key block's K and V lines toward this WGP
        if (kb0 + 64 < NKEY) {
            __builtin_prefetch(kb + (size_t)(kb0 + 64 + scw * 16 + lrow) * DDIM, 0, 1);
            __builtin_prefetch(vb + (size_t)(cb + lrow) * NKEY + kb0 + 64, 0, 1);
        }

        // ---- S = Q @ K^T (one 16x16 tile per wave, K-dim = 512) ----
        v8f sacc;
        #pragma unroll
        for (int j = 0; j < 8; ++j) sacc[j] = 0.0f;
        const _Float16* krow = kb + (size_t)(kb0 + scw * 16 + lrow) * DDIM;
        #pragma unroll
        for (int ks = 0; ks < DDIM; ks += 32) {
            F16Frag a, kf;
            a.h[0]  = *(const h8*)(qrow + ks + hisel * 16);
            a.h[1]  = *(const h8*)(qrow + ks + hisel * 16 + 8);
            kf.h[0] = *(const h8*)(krow + ks + hisel * 16);
            kf.h[1] = *(const h8*)(krow + ks + hisel * 16 + 8);
            sacc = wmma_f16<false, false>(a, kf, sacc);
        }
        #pragma unroll
        for (int j = 0; j < 8; ++j)
            S_lds[srw * 16 + hisel * 8 + j][scw * 16 + lrow] = sacc[j];
        __syncthreads();

        // ---- online softmax row stats (one thread per query row) ----
        if (tid < 32) {
            float mo = m_s[tid];
            float mx = mo;
            #pragma unroll 8
            for (int c = 0; c < 64; ++c) mx = fmaxf(mx, S_lds[tid][c]);
            float f = __expf(mo - mx);     // 0 on first block (mo = -3e38)
            m_s[tid] = mx;
            f_s[tid] = f;
            l_s[tid] *= f;
        }
        __syncthreads();

        // ---- P = exp(S - m), f16 into LDS; row-sum via LDS float atomics ----
        {
            int r  = tid >> 3;             // 8 threads per row
            int c0 = (tid & 7) * 8;
            float mrow = m_s[r];
            float lsum = 0.0f;
            #pragma unroll
            for (int cc = 0; cc < 8; ++cc) {
                float p = __expf(S_lds[r][c0 + cc] - mrow);
                P_lds[r][c0 + cc] = (_Float16)p;
                lsum += p;
            }
            atomicAdd(&l_s[r], lsum);
        }
        __syncthreads();

        // ---- rescale running output accumulators ----
        #pragma unroll
        for (int r = 0; r < 2; ++r) {
            #pragma unroll
            for (int j = 0; j < 8; ++j) {
                float f = f_s[r * 16 + hisel * 8 + j];
                #pragma unroll
                for (int ct = 0; ct < 4; ++ct) oacc[r][ct][j] *= f;
            }
        }

        // ---- O += P @ V  (K-dim = 64 keys -> 2 WMMA steps) ----
        #pragma unroll
        for (int s = 0; s < 2; ++s) {
            F16Frag pa[2];
            #pragma unroll
            for (int r = 0; r < 2; ++r) {
                const _Float16* pp = &P_lds[r * 16 + lrow][s * 32 + hisel * 16];
                pa[r].h[0] = *(const h8*)(pp);
                pa[r].h[1] = *(const h8*)(pp + 8);
            }
            F16Frag vf[4];
            #pragma unroll
            for (int ct = 0; ct < 4; ++ct) {
                const _Float16* vp =
                    vb + (size_t)(cb + ct * 16 + lrow) * NKEY + kb0 + s * 32 + hisel * 16;
                vf[ct].h[0] = *(const h8*)(vp);
                vf[ct].h[1] = *(const h8*)(vp + 8);
            }
            // r-outer: 4 consecutive WMMAs share pa[r] -> reuse_a hints
            #pragma unroll
            for (int r = 0; r < 2; ++r) {
                oacc[r][0] = wmma_f16<true,  false>(pa[r], vf[0], oacc[r][0]);
                oacc[r][1] = wmma_f16<true,  false>(pa[r], vf[1], oacc[r][1]);
                oacc[r][2] = wmma_f16<true,  false>(pa[r], vf[2], oacc[r][2]);
                oacc[r][3] = wmma_f16<false, false>(pa[r], vf[3], oacc[r][3]);
            }
        }
        __syncthreads();
    }

    // ---- finalize: divide by l, write f32 output ----
    if (tid < 32) f_s[tid] = 1.0f / l_s[tid];
    __syncthreads();
    #pragma unroll
    for (int r = 0; r < 2; ++r) {
        #pragma unroll
        for (int ct = 0; ct < 4; ++ct) {
            #pragma unroll
            for (int j = 0; j < 8; ++j) {
                int row = r * 16 + hisel * 8 + j;
                int col = cb + ct * 16 + lrow;
                ob[(size_t)(q0 + row) * DDIM + col] = oacc[r][ct][j] * f_s[row];
            }
        }
    }
}

// ---------------------------------------------------------------------------
extern "C" void kernel_launch(void* const* d_in, const int* in_sizes, int n_in,
                              void* d_out, int out_size, void* d_ws, size_t ws_size,
                              hipStream_t stream) {
    (void)in_sizes; (void)n_in; (void)out_size; (void)ws_size;
    const float* img = (const float*)d_in[0];
    const float* ts  = (const float*)d_in[1];
    const float* Wq  = (const float*)d_in[2];
    const float* bq  = (const float*)d_in[3];
    const float* Wk  = (const float*)d_in[4];
    const float* bk  = (const float*)d_in[5];
    const float* Wv  = (const float*)d_in[6];
    const float* bv  = (const float*)d_in[7];
    float* out = (float*)d_out;

    _Float16* qh = (_Float16*)d_ws;                       // [B*NQ][D] f16
    _Float16* kh = qh + (size_t)BATCH * NQ * DDIM;        // [B*NK][D] f16
    _Float16* vt = kh + (size_t)BATCH * NKEY * DDIM;      // [B][D][NK] f16

    dim3 pgrid(BATCH * NQ / 128, DDIM / 64);
    proj_f16_kernel<0><<<pgrid, 256, 0, stream>>>(img, Wq, bq, qh);
    proj_f16_kernel<0><<<pgrid, 256, 0, stream>>>(ts,  Wk, bk, kh);
    proj_f16_kernel<1><<<pgrid, 256, 0, stream>>>(ts,  Wv, bv, vt);

    dim3 agrid(NQ / 32, BATCH);
    attn_kernel<<<agrid, 256, 0, stream>>>(qh, kh, vt, out);
}
